// AFM_51101520888212
// MI455X (gfx1250) — compile-verified, matrix operand order checked by reading, not run on hardware
//
#include <hip/hip_runtime.h>
#include <math.h>

typedef __attribute__((ext_vector_type(16))) _Float16 v16h;
typedef __attribute__((ext_vector_type(8)))  float    v8f;

#define NF 39
#define NE 16
#define NA 16
#define NP 741          // NF*(NF-1)/2
#define NT 47           // ceil(NP/16)
#define WAVES 8
#define ESTR 20         // emb row stride (floats): 16B-aligned rows, conflict-free
#define ASTR 20         // att tile row stride (floats)

// wave-local ordering fence: LDS ops within a wave are in-order on CDNA5;
// we only need to stop the compiler from reordering across the relayout.
#define WAVE_FENCE() do { asm volatile("" ::: "memory"); \
                          __builtin_amdgcn_wave_barrier(); \
                          asm volatile("" ::: "memory"); } while (0)

__global__ __launch_bounds__(256) void afm_fused_kernel(
    const int*   __restrict__ feat_index,   // [B,39]
    const float* __restrict__ feat_value,   // [B,39]
    const float* __restrict__ w1,           // [V,1]
    const float* __restrict__ emb_table,    // [V,16]
    const float* __restrict__ bias,         // [1]
    const float* __restrict__ attn_w,       // [16,16]
    const float* __restrict__ attn_b,       // [16]
    const float* __restrict__ proj_h,       // [16,1]
    const float* __restrict__ proj_p,       // [16,1]
    float*       __restrict__ out,          // [B]
    int B)
{
    __shared__ int   pairS[NP];                  // packed (i<<16)|j, block-shared
    __shared__ float embS[WAVES][NF * ESTR];     // per-wave scaled embeddings
    __shared__ float scS[WAVES][NP];             // per-wave pair scores / exp weights
    __shared__ float attS[WAVES][16 * ASTR];     // per-wave att tile relayout buffer

    const int tid  = threadIdx.x;
    const int wave = tid >> 5;
    const int lane = tid & 31;                   // wave32
    const int lh   = lane & 15;                  // lane within 16-lane half
    const int grp  = lane >> 4;                  // 0 or 1

    // ---- build pair (i,j) unranking table once per block ----
    for (int p = tid; p < NP; p += blockDim.x) {
        int i = 0, rem = p;
        while (rem >= (NF - 1 - i)) { rem -= (NF - 1 - i); ++i; }
        pairS[p] = (i << 16) | (i + 1 + rem);
    }
    __syncthreads();

    int b = blockIdx.x * WAVES + wave;
    if (b >= B) b = B - 1;                       // clamp (B is a multiple of 8)
    const int*   idxRow = feat_index + b * NF;
    const float* fvRow  = feat_value + b * NF;
    float* emb = embS[wave];
    float* sc  = scS[wave];
    float* att = attS[wave];

    // ---- gather embeddings scaled by feature value (float4 = b128 traffic) ----
    for (int k = lane; k < NF * 4; k += 32) {
        const int f = k >> 2, q = k & 3;
        const float fv = fvRow[f];
        const float4 v = ((const float4*)(emb_table + (size_t)idxRow[f] * NE))[q];
        float4 s; s.x = v.x * fv; s.y = v.y * fv; s.z = v.z * fv; s.w = v.w * fv;
        *(float4*)(emb + f * ESTR + q * 4) = s;
    }

    // ---- first-order term ----
    float yf = 0.f;
    for (int f = lane; f < NF; f += 32)
        yf += w1[idxRow[f]] * fvRow[f];
    #pragma unroll
    for (int m = 16; m >= 1; m >>= 1) yf += __shfl_xor(yf, m, 32);

    // ---- static B fragments ----
    // B layout (f16, 32x16): N = lh; grp0 halves0-7 = K0-7, grp1 halves0-7 = K8-15.
    v16h bfW;    // attention_w: rows K=0..15 real, K=16..31 zero-pad
    v16h bfH;    // proj_h in column 0 only
    #pragma unroll
    for (int h = 0; h < 8; ++h) {
        const int k = grp * 8 + h;
        bfW[h]     = (_Float16)attn_w[k * NA + lh];
        bfW[8 + h] = (_Float16)0.f;
        bfH[h]     = (lh == 0) ? (_Float16)proj_h[k] : (_Float16)0.f;
        bfH[8 + h] = (_Float16)0.f;
    }
    const float ab = attn_b[lh];

    WAVE_FENCE();

    // ================= pass 1: scores via two chained WMMAs per tile ========
    for (int t = 0; t < NT; ++t) {
        // A1 = bi tile: row M=lh (pair t*16+lh), K = e (grp0: 0-7, grp1: 8-15)
        const int p = t * 16 + lh;
        v16h a1;
        if (p < NP) {
            const int pr = pairS[p];
            const float* ri = emb + (pr >> 16) * ESTR + grp * 8;
            const float* rj = emb + (pr & 0xFFFF) * ESTR + grp * 8;
            const float4 i0 = *(const float4*)(ri);
            const float4 i1 = *(const float4*)(ri + 4);
            const float4 j0 = *(const float4*)(rj);
            const float4 j1 = *(const float4*)(rj + 4);
            a1[0] = (_Float16)(i0.x * j0.x); a1[1] = (_Float16)(i0.y * j0.y);
            a1[2] = (_Float16)(i0.z * j0.z); a1[3] = (_Float16)(i0.w * j0.w);
            a1[4] = (_Float16)(i1.x * j1.x); a1[5] = (_Float16)(i1.y * j1.y);
            a1[6] = (_Float16)(i1.z * j1.z); a1[7] = (_Float16)(i1.w * j1.w);
            #pragma unroll
            for (int h = 8; h < 16; ++h) a1[h] = (_Float16)0.f;   // K pad
        } else {
            #pragma unroll
            for (int h = 0; h < 16; ++h) a1[h] = (_Float16)0.f;
        }

        v8f c1 = {};
        c1 = __builtin_amdgcn_wmma_f32_16x16x32_f16(
                 false, a1, false, bfW, (short)0, c1, false, false);

        // att = relu(c1 + b); D layout: VGPR r -> M = grp*8+r, N = lh
        #pragma unroll
        for (int r = 0; r < 8; ++r)
            att[(grp * 8 + r) * ASTR + lh] = fmaxf(c1[r] + ab, 0.f);
        WAVE_FENCE();

        // A2 = att tile in A layout: row M=lh, K = grp*8 + h
        const float* ar = att + lh * ASTR + grp * 8;
        const float4 t0 = *(const float4*)(ar);
        const float4 t1 = *(const float4*)(ar + 4);
        v16h a2;
        a2[0] = (_Float16)t0.x; a2[1] = (_Float16)t0.y;
        a2[2] = (_Float16)t0.z; a2[3] = (_Float16)t0.w;
        a2[4] = (_Float16)t1.x; a2[5] = (_Float16)t1.y;
        a2[6] = (_Float16)t1.z; a2[7] = (_Float16)t1.w;
        #pragma unroll
        for (int h = 8; h < 16; ++h) a2[h] = (_Float16)0.f;
        WAVE_FENCE();

        v8f c2 = {};
        c2 = __builtin_amdgcn_wmma_f32_16x16x32_f16(
                 false, a2, false, bfH, (short)0, c2, false, false);

        // score column 0 lives in lanes lh==0: rows M = grp*8 + r
        if (lh == 0) {
            #pragma unroll
            for (int r = 0; r < 8; ++r) {
                const int pp = t * 16 + grp * 8 + r;
                if (pp < NP) sc[pp] = c2[r];
            }
        }
    }
    WAVE_FENCE();

    // ================= softmax over 741 pair scores =========================
    float mx = -1e30f;
    for (int p = lane; p < NP; p += 32) mx = fmaxf(mx, sc[p]);
    #pragma unroll
    for (int m = 16; m >= 1; m >>= 1) mx = fmaxf(mx, __shfl_xor(mx, m, 32));
    float se = 0.f;
    for (int p = lane; p < NP; p += 32) {
        const float e2 = __expf(sc[p] - mx);
        sc[p] = e2;                              // unnormalized weights
        se += e2;
    }
    #pragma unroll
    for (int m = 16; m >= 1; m >>= 1) se += __shfl_xor(se, m, 32);
    const float inv = 1.f / se;
    WAVE_FENCE();

    // ========= pass 2: pooled = BI^T @ w via C-accumulating WMMA ============
    v8f c3 = {};
    for (int t = 0; t < NT; ++t) {
        // A3 = bi^T tile: row M = lh = e; K = pair (grp0: 0-7, grp1: 8-15)
        v16h a3;
        #pragma unroll
        for (int h = 0; h < 8; ++h) {
            const int p = t * 16 + grp * 8 + h;
            float biv = 0.f;
            if (p < NP) {
                const int pr = pairS[p];
                biv = emb[(pr >> 16) * ESTR + lh] * emb[(pr & 0xFFFF) * ESTR + lh];
            }
            a3[h]     = (_Float16)biv;
            a3[8 + h] = (_Float16)0.f;
        }
        // B3 = exp-weights in column 0
        v16h b3;
        #pragma unroll
        for (int h = 0; h < 8; ++h) {
            const int p = t * 16 + grp * 8 + h;
            b3[h]     = (lh == 0 && p < NP) ? (_Float16)sc[p] : (_Float16)0.f;
            b3[8 + h] = (_Float16)0.f;
        }
        c3 = __builtin_amdgcn_wmma_f32_16x16x32_f16(
                 false, a3, false, b3, (short)0, c3, false, false);
    }

    // pooled[e] in column 0: lane 0 holds e=0..7, lane 16 holds e=8..15
    float awp = 0.f;
    if (lh == 0) {
        #pragma unroll
        for (int r = 0; r < 8; ++r) awp += c3[r] * proj_p[grp * 8 + r];
    }
    awp += __shfl_xor(awp, 16, 32);

    if (lane == 0) {
        const float y = bias[0] + yf + awp * inv;
        out[b] = 1.f / (1.f + __expf(-y));
    }
}

extern "C" void kernel_launch(void* const* d_in, const int* in_sizes, int n_in,
                              void* d_out, int out_size, void* d_ws, size_t ws_size,
                              hipStream_t stream) {
    const int*   feat_index = (const int*)  d_in[0];
    const float* feat_value = (const float*)d_in[1];
    const float* w1         = (const float*)d_in[2];
    const float* emb_table  = (const float*)d_in[3];
    const float* bias       = (const float*)d_in[4];
    const float* attn_w     = (const float*)d_in[5];
    const float* attn_b     = (const float*)d_in[6];
    const float* proj_h     = (const float*)d_in[7];
    const float* proj_p     = (const float*)d_in[8];
    float* out = (float*)d_out;

    const int B = in_sizes[0] / NF;
    const int blocks = (B + WAVES - 1) / WAVES;
    afm_fused_kernel<<<blocks, 256, 0, stream>>>(
        feat_index, feat_value, w1, emb_table, bias,
        attn_w, attn_b, proj_h, proj_p, out, B);
}
